// SequentialFeedForward_84370337562848
// MI455X (gfx1250) — matrix operand order repaired and checked by reading, not compiled, not run on hardware
//
#include <hip/hip_runtime.h>
#include <hip/hip_bf16.h>

#define WS 16
#define NUM_HIDDEN 3
#define WIDTH 1024
#define ALPHA 21
#define IN_ONEHOT ((WS + 1) * ALPHA)   /* 357 */
#define T_LEN 16384

typedef __attribute__((ext_vector_type(16))) __bf16       v16bf;
typedef __attribute__((ext_vector_type(8)))  float        v8f;
typedef __attribute__((ext_vector_type(4)))  unsigned int u32x4;

union ABu { u32x4 q[2]; v16bf v; };

// ---------------------------------------------------------------------------
// Kernel 1: repack Wh (fp32 [3][1024][1024], out[n] = sum_k h[k]*Wh[l][k][n])
// into bf16 WMMA B-tiles. Tile (l, nt, kt) is 32x16 (KxN), stored as
// [lane 0..31][elem 0..15] with   n = nt*16 + (lane&15),
//                                k = kt*32 + (lane>>4)*16 + elem.
// Tile order: [l][nt][kt] so the sequential kernel streams kt contiguously.
// ---------------------------------------------------------------------------
__global__ void pack_wh_kernel(const float* __restrict__ Wh,
                               __bf16* __restrict__ whp) {
    int gtid = blockIdx.x * blockDim.x + threadIdx.x;
    int tile = gtid >> 5;          // 3*64*32 = 6144 tiles
    int lane = gtid & 31;
    int layer = tile / (64 * 32);
    int rem   = tile % (64 * 32);
    int nt = rem >> 5;
    int kt = rem & 31;
    int n  = nt * 16 + (lane & 15);
    int kb = kt * 32 + (lane >> 4) * 16;
    const float* src = Wh + (size_t)layer * WIDTH * WIDTH;
    __bf16* dst = whp + ((size_t)tile * 32 + lane) * 16;
#pragma unroll
    for (int e = 0; e < 16; ++e)
        dst[e] = (__bf16)src[(size_t)(kb + e) * WIDTH + n];
}

// ---------------------------------------------------------------------------
// Kernel 2: pre[t][col] = b0[col] + sum_{j=0..16} W0[row_j][col]
// where row_j = j*21 + x[t+j-16], or the all-ones padding contribution
// (sum of the 21 rows of block j) when t+j-16 < 0.
// ---------------------------------------------------------------------------
__global__ __launch_bounds__(1024) void pre_kernel(const int* __restrict__ x,
                                                   const float* __restrict__ W0,
                                                   const float* __restrict__ b0,
                                                   float* __restrict__ pre) {
    int t = blockIdx.x;
    int col = threadIdx.x;
    float acc = b0[col];
#pragma unroll
    for (int j = 0; j <= WS; ++j) {
        int tj = t + j - WS;
        if (tj >= 0) {
            int row = j * ALPHA + x[tj];
            acc += W0[(size_t)row * WIDTH + col];
        } else {
            for (int c = 0; c < ALPHA; ++c)
                acc += W0[(size_t)(j * ALPHA + c) * WIDTH + col];
        }
    }
    pre[(size_t)t * WIDTH + col] = acc;
}

// ---------------------------------------------------------------------------
// Kernel 3: sequential recurrence. One workgroup, 1024 threads = 32 waves.
// Per step: layer0 (VALU, rank-16 update on precomputed pre[t]),
// 3 x (1024x1024 bf16 matvec via WMMA, A = activation broadcast to all rows),
// output dot product + ring update.
// ---------------------------------------------------------------------------
__global__ __launch_bounds__(1024, 1)
void seq_kernel(const float* __restrict__ W0,
                const float* __restrict__ bh,
                const float* __restrict__ Wo,
                const float* __restrict__ bo,
                const float* __restrict__ pre,
                const void*  __restrict__ whp_raw,
                float* __restrict__ out) {
    const __bf16* __restrict__ whp = (const __bf16*)whp_raw;
    const int tid  = threadIdx.x;
    const int wave = tid >> 5;
    const int lane = tid & 31;

    __shared__ __bf16 WyL[WS * WIDTH];                       // 32 KB
    __shared__ float  WoL[WIDTH];                            //  4 KB
    __shared__ float  bhL[NUM_HIDDEN * WIDTH];               // 12 KB
    __shared__ float  ring[WS];
    __shared__ __align__(16) __bf16 hbuf[2][WIDTH];          //  4 KB

    // Preload constants into LDS.
#pragma unroll
    for (int k = 0; k < WS; ++k)
        WyL[k * WIDTH + tid] = (__bf16)W0[(size_t)(IN_ONEHOT + k) * WIDTH + tid];
    WoL[tid] = Wo[tid];
#pragma unroll
    for (int l = 0; l < NUM_HIDDEN; ++l)
        bhL[l * WIDTH + tid] = bh[l * WIDTH + tid];
    if (tid < WS) ring[tid] = -10.0f;
    const float bo_s = bo[0];
    __syncthreads();

    const int nt0 = wave * 2;
    const int nt1 = wave * 2 + 1;
    const int hh8 = (lane >> 4) * 8;

    for (int t = 0; t < T_LEN; ++t) {
        // ---- layer 0: pre[t] + y_win @ Wy, relu, -> hbuf[0] (bf16) ----
        {
            float acc = pre[(size_t)t * WIDTH + tid];
#pragma unroll
            for (int k = 0; k < WS; ++k)
                acc += ring[(t + k) & (WS - 1)] * (float)WyL[k * WIDTH + tid];
            hbuf[0][tid] = (__bf16)fmaxf(acc, 0.0f);
            if (t + 1 < T_LEN)
                __builtin_prefetch(&pre[(size_t)(t + 1) * WIDTH + tid], 0, 1);
        }
        __syncthreads();

        // ---- hidden layers: WMMA matvecs ----
        for (int l = 0; l < NUM_HIDDEN; ++l) {
            const int src = l & 1;        // 0,1,0
            const int dst = src ^ 1;      // 1,0,1
            v8f c0 = {};
            v8f c1 = {};
            const __bf16* bp0 = whp + ((size_t)(l * 64 + nt0) * 32) * 512 + lane * 16;
            const __bf16* bp1 = whp + ((size_t)(l * 64 + nt1) * 32) * 512 + lane * 16;
#pragma unroll 2
            for (int kt = 0; kt < 32; ++kt) {
                ABu a, bb0, bb1;
                // A: activation chunk broadcast to all 16 rows.
                // lanes 0-15 need h[kt*32 + 0..15];
                // lanes 16-31 need h[kt*32 + 16..31].
                a.q[0] = *(const u32x4*)&hbuf[src][kt * 32 + 2 * hh8];
                a.q[1] = *(const u32x4*)&hbuf[src][kt * 32 + 2 * hh8 + 8];
                const u32x4* p0 = (const u32x4*)(bp0 + (size_t)kt * 512);
                const u32x4* p1 = (const u32x4*)(bp1 + (size_t)kt * 512);
                bb0.q[0] = p0[0]; bb0.q[1] = p0[1];
                bb1.q[0] = p1[0]; bb1.q[1] = p1[1];
                c0 = __builtin_amdgcn_wmma_f32_16x16x32_bf16(
                        false, a.v, false, bb0.v, (short)0, c0, false, false);
                c1 = __builtin_amdgcn_wmma_f32_16x16x32_bf16(
                        false, a.v, false, bb1.v, (short)0, c1, false, false);
            }
            // Every row of D equals the matvec result; lanes 0-15, acc VGPR0
            // hold out[n] for n = nt*16 + lane.
            if (lane < 16) {
                float v0 = c0[0] + bhL[l * WIDTH + nt0 * 16 + lane];
                hbuf[dst][nt0 * 16 + lane] = (__bf16)fmaxf(v0, 0.0f);
                float v1 = c1[0] + bhL[l * WIDTH + nt1 * 16 + lane];
                hbuf[dst][nt1 * 16 + lane] = (__bf16)fmaxf(v1, 0.0f);
            }
            __syncthreads();
        }

        // ---- output: relu(dot(h, Wo) + bo), ring update ----
        if (wave == 0) {
            float s = 0.0f;
#pragma unroll
            for (int j = 0; j < 32; ++j) {
                int idx = lane + j * 32;
                s += (float)hbuf[1][idx] * WoL[idx];
            }
#pragma unroll
            for (int off = 16; off > 0; off >>= 1)
                s += __shfl_down(s, off, 32);
            if (lane == 0) {
                float y = fmaxf(s + bo_s, 0.0f);
                out[t] = y;
                ring[t & (WS - 1)] = y;
            }
        }
        __syncthreads();
    }
}

// ---------------------------------------------------------------------------
extern "C" void kernel_launch(void* const* d_in, const int* in_sizes, int n_in,
                              void* d_out, int out_size, void* d_ws, size_t ws_size,
                              hipStream_t stream) {
    const int*   x  = (const int*)d_in[0];
    const float* W0 = (const float*)d_in[1];
    const float* b0 = (const float*)d_in[2];
    const float* Wh = (const float*)d_in[3];
    const float* bh = (const float*)d_in[4];
    const float* Wo = (const float*)d_in[5];
    const float* bo = (const float*)d_in[6];
    float* out = (float*)d_out;

    char* ws = (char*)d_ws;
    float*  pre = (float*)ws;                                   // 64 MB
    __bf16* whp = (__bf16*)(ws + (size_t)T_LEN * WIDTH * 4);    //  6 MB

    // 6144 tiles * 32 lanes = 196608 threads
    pack_wh_kernel<<<768, 256, 0, stream>>>(Wh, whp);
    pre_kernel<<<T_LEN, 1024, 0, stream>>>(x, W0, b0, pre);
    seq_kernel<<<1, 1024, 0, stream>>>(W0, bh, Wo, bo, pre, (const void*)whp, out);
}
